// Head_31696858644931
// MI455X (gfx1250) — matrix operand order repaired and checked by reading, not compiled, not run on hardware
//
#include <hip/hip_runtime.h>
#include <hip/hip_bf16.h>

// ---------------------------------------------------------------------------
// Fused causal single-head attention for gfx1250 (MI455X, wave32, WMMA).
//   ref: q,k,v = x@W*, scores = qk^T/sqrt(64) + causal mask, out = softmax@v
// Pipeline:
//   K0: convert Wq/Wk/Wv (f32 [1024,64]) -> Wt bf16 [3][64][1024] (transposed)
//   K1: QKV projection with v_wmma_f32_16x16x32_bf16. Ping-pong A (d-loop
//       unrolled x2) and ping-pong B (chain unrolled x2) buffers: loads land
//       directly in their WMMA source registers (no rotation copies), tail
//       peeled so the pipelined body has no branches.
//   K2: flash-attention-2 (online softmax) with WMMA for QK^T and PV.
//       K tile staged into LDS by the Tensor Data Mover (s_wait_tensorcnt
//       completion) when available, V^T staged manually. Causal mask computed
//       analytically (the 64MB mask input is never read).
// ---------------------------------------------------------------------------

typedef __attribute__((ext_vector_type(16))) __bf16 v16bf;
typedef __attribute__((ext_vector_type(8)))  float  v8f;

static constexpr int Bn  = 4;
static constexpr int Tn  = 4096;
static constexpr int Dn  = 1024;
static constexpr int Hn  = 64;
static constexpr int BTn = Bn * Tn;          // 16384 flattened rows

union AFrag { unsigned int u[8]; v16bf v; };   // 16 bf16 = 8 dwords
union BFrag { uint4 q[2];        v16bf v; };   // 16 bf16 = 2 x b128

// f32 -> bf16, round-half-up (1 add + shift; cheap VALU)
__device__ __forceinline__ unsigned short f2bf(float f) {
    return (unsigned short)((__float_as_uint(f) + 0x8000u) >> 16);
}
// pack two f32 -> packed bf16 pair. One v_perm_b32 selects the two high
// halves after rounding bias: 3 VALU total.
__device__ __forceinline__ unsigned int pack_bf2(float lo, float hi) {
#if __has_builtin(__builtin_amdgcn_cvt_pk_bf16_f32)
    auto t = __builtin_amdgcn_cvt_pk_bf16_f32(lo, hi);
    return __builtin_bit_cast(unsigned int, t);
#else
    unsigned int a = __float_as_uint(lo) + 0x8000u;
    unsigned int b = __float_as_uint(hi) + 0x8000u;
    // dest bytes {b3,b2, a3,a2} -> {hi.bf16, lo.bf16}
    return __builtin_amdgcn_perm(b, a, 0x07060302u);
#endif
}
__device__ __forceinline__ v8f zero8() {
    v8f z;
#pragma unroll
    for (int i = 0; i < 8; ++i) z[i] = 0.0f;
    return z;
}
// 16-bit A-matrix 16x32 layout (contiguous k order): VGPR v holds K pair starting at:
//   half*16 + 2*v
__device__ __forceinline__ int a_kbase(int v, int half) {
    return half * 16 + 2 * v;
}

// ---------------------------------------------------------------------------
// Tensor Data Mover: 2D tile load (32 keys x 64 bf16, row-major) into LDS.
// Descriptor packed per CDNA5 ISA.
// ---------------------------------------------------------------------------
#if __has_builtin(__builtin_amdgcn_tensor_load_to_lds) && \
    __has_builtin(__builtin_amdgcn_s_wait_tensorcnt)
#define USE_TDM 1
typedef unsigned int u32x4 __attribute__((ext_vector_type(4)));
typedef int          i32x4 __attribute__((ext_vector_type(4)));
typedef int          i32x8 __attribute__((ext_vector_type(8)));

__device__ __forceinline__ void tdm_load_k_tile(const unsigned short* gsrc,
                                                unsigned short* lds_dst) {
    unsigned long long ga = (unsigned long long)gsrc;
    // generic LDS addresses carry the LDS byte offset in bits [31:0]
    unsigned int laddr = (unsigned int)(unsigned long long)(void*)lds_dst;
    u32x4 g0;
    g0[0] = 1u;                                          // count=1, user D#
    g0[1] = laddr;                                       // lds_addr
    g0[2] = (unsigned int)(ga & 0xFFFFFFFFu);            // global_addr[31:0]
    g0[3] = (unsigned int)((ga >> 32) & 0x01FFFFFFu)     // global_addr[56:32]
            | (2u << 30);                                // type = 2 (image)
    i32x8 g1;
    g1[0] = (int)(1u << 16);       // wg_mask=0, data_size=1 (2 bytes)
    g1[1] = (int)(64u << 16);      // tensor_dim0 = 64 (lo16 in [31:16])
    g1[2] = (int)(4096u << 16);    // tensor_dim0 hi=0, tensor_dim1 lo16 = 4096
    g1[3] = (int)(64u << 16);      // tensor_dim1 hi=0, tile_dim0 = 64
    g1[4] = 32;                    // tile_dim1 = 32, tile_dim2 = 0
    g1[5] = 64;                    // tensor_dim0_stride[31:0] = 64
    g1[6] = 0;                     // stride hi / dim1_stride lo
    g1[7] = 0;
    i32x4 z4 = {0, 0, 0, 0};
#if __clang_major__ >= 23
    i32x8 z8 = {0, 0, 0, 0, 0, 0, 0, 0};
    __builtin_amdgcn_tensor_load_to_lds(g0, g1, z4, z4, z8, 0);
#else
    __builtin_amdgcn_tensor_load_to_lds(g0, g1, z4, z4, 0);
#endif
}
#endif

// ---------------------------------------------------------------------------
// K0: Wt[p][h][d] = bf16(W_p[d][h])   (p in {q,k,v})
// ---------------------------------------------------------------------------
__global__ void attn_wcvt_kernel(const float* __restrict__ Wq,
                                 const float* __restrict__ Wk,
                                 const float* __restrict__ Wv,
                                 unsigned short* __restrict__ Wt) {
    int idx = blockIdx.x * blockDim.x + threadIdx.x;
    const int total = 3 * Hn * Dn;
    if (idx >= total) return;
    int p  = idx / (Hn * Dn);
    int r  = idx % (Hn * Dn);
    int h  = r / Dn;
    int dd = r % Dn;
    const float* W = (p == 0) ? Wq : ((p == 1) ? Wk : Wv);
    Wt[idx] = f2bf(W[dd * Hn + h]);
}

// ---------------------------------------------------------------------------
// K1: QKV GEMM. 128 threads = 4 waves; wave w owns 16 rows of x and computes
//     all 64 output cols for all three projections (12 accum tiles / wave).
// ---------------------------------------------------------------------------
__global__ void __launch_bounds__(128, 1)
attn_qkv_kernel(const float* __restrict__ x,
                const unsigned short* __restrict__ Wt,
                unsigned short* __restrict__ qkv) {
    const int wave = threadIdx.x >> 5;
    const int lane = threadIdx.x & 31;
    const int half = lane >> 4;
    const int l16  = lane & 15;

    const int row0 = (blockIdx.x * 4 + wave) * 16;   // 16-row tile base
    const int rowA = row0 + l16;                     // A-layout: m = lane%16

    v8f acc[12];
#pragma unroll
    for (int i = 0; i < 12; ++i) acc[i] = zero8();

    const float* xrow = x + (long)rowA * Dn;
    const unsigned short* wl = Wt + (long)l16 * Dn + half * 16;

    auto buildA = [&](int d) {
        AFrag a;
#pragma unroll
        for (int v = 0; v < 8; ++v) {
            int kk = a_kbase(v, half);
            float2 f = *(const float2*)(xrow + d + kk);
            a.u[v] = pack_bf2(f.x, f.y);
        }
        return a;
    };
    auto loadB = [&](int i, int d) {   // i = p*4 + nt
        BFrag b;
        const uint4* wp = (const uint4*)(wl +
            (long)((i >> 2) * Hn + (i & 3) * 16) * Dn + d);
        b.q[0] = wp[0];
        b.q[1] = wp[1];
        return b;
    };

    AFrag a0, a1;
    BFrag b0, b1;

    // 12-tile WMMA chain for one 32-deep k-step. b0/b1 ping-pong so every
    // load lands directly in its (even-aligned) WMMA source registers.
    auto chain = [&](const AFrag& a, int d, int dnext, bool pf) {
#pragma unroll
        for (int i = 0; i < 12; i += 2) {
            b1 = loadB(i + 1, d);
            acc[i] = __builtin_amdgcn_wmma_f32_16x16x32_bf16(
                false, a.v, false, b0.v, (short)0, acc[i], false, false);
            if (i + 2 < 12)  b0 = loadB(i + 2, d);
            else if (pf)     b0 = loadB(0, dnext);
            acc[i + 1] = __builtin_amdgcn_wmma_f32_16x16x32_bf16(
                false, a.v, false, b1.v, (short)0, acc[i + 1], false, false);
        }
    };

    a0 = buildA(0);
    b0 = loadB(0, 0);
    // main loop: two k-steps per iteration, A ping-pongs between a0/a1
    for (int dd = 0; dd < Dn - 64; dd += 64) {
        a1 = buildA(dd + 32);
        chain(a0, dd, dd + 32, true);
        a0 = buildA(dd + 64);
        chain(a1, dd + 32, dd + 64, true);
    }
    // peeled tail: dd = Dn-64 (no out-of-range prefetch, no branches)
    a1 = buildA(Dn - 32);
    chain(a0, Dn - 64, Dn - 32, true);
    chain(a1, Dn - 32, 0, false);

    // C/D layout: VGPR r, lanes 0-15 -> M=r, lanes 16-31 -> M=r+8
#pragma unroll
    for (int p = 0; p < 3; ++p)
#pragma unroll
        for (int nt = 0; nt < 4; ++nt)
#pragma unroll
            for (int r = 0; r < 8; ++r) {
                int row = row0 + r + half * 8;
                int col = nt * 16 + l16;
                qkv[(long)p * BTn * Hn + (long)row * Hn + col] =
                    f2bf(acc[p * 4 + nt][r]);
            }
}

// ---------------------------------------------------------------------------
// K2: flash attention. Block = 128 thr (4 waves), covers 64 q rows of one
//     batch. Key loop in blocks of 32; K tile (TDM) + V^T tile staged in LDS.
// ---------------------------------------------------------------------------
__global__ void attn_fa_kernel(const unsigned short* __restrict__ qkv,
                               float* __restrict__ out) {
    const int wave = threadIdx.x >> 5;
    const int lane = threadIdx.x & 31;
    const int half = lane >> 4;
    const int l16  = lane & 15;
    const int tid  = threadIdx.x;

    const int b     = blockIdx.x >> 6;        // batch
    const int qtile = blockIdx.x & 63;        // 64-row q block within batch
    const int q0    = qtile * 64;
    const int qw0   = q0 + wave * 16;         // this wave's 16 q rows

    const unsigned short* Qb = qkv + 0L * BTn * Hn + (long)b * Tn * Hn;
    const unsigned short* Kb = qkv + 1L * BTn * Hn + (long)b * Tn * Hn;
    const unsigned short* Vb = qkv + 2L * BTn * Hn + (long)b * Tn * Hn;

    __shared__ __align__(16) unsigned short Klds[32 * 64];   // [key][h]
    __shared__ __align__(16) unsigned short Vlds[64 * 32];   // [h][key] (V^T)
    __shared__ __align__(16) unsigned short Plds[4][16 * 32];// per-wave P tile

    // Preload Q fragments (A layout), kept in registers for the whole loop.
    AFrag aQ[2];
#pragma unroll
    for (int ks = 0; ks < 2; ++ks)
#pragma unroll
        for (int v = 0; v < 8; ++v) {
            int h = ks * 32 + a_kbase(v, half);
            aQ[ks].u[v] = *(const unsigned int*)(Qb + (long)(qw0 + l16) * Hn + h);
        }

    v8f   o[4];
    float mrow[8], lrow[8];
#pragma unroll
    for (int nt = 0; nt < 4; ++nt) o[nt] = zero8();
#pragma unroll
    for (int r = 0; r < 8; ++r) { mrow[r] = -INFINITY; lrow[r] = 0.0f; }

    const float inv_scale = 0.125f;           // 1/sqrt(64)
    const int nkb = (q0 + 64) / 32;           // causal: keys 0 .. q0+63

    for (int kb = 0; kb < nkb; ++kb) {
        const int k0 = kb * 32;
        __syncthreads();                      // prior iter done reading LDS

        // ---- Stage K block [key][h] ----
#ifdef USE_TDM
        if (wave == 0) {
            tdm_load_k_tile(Kb + (long)k0 * Hn, Klds);
            __builtin_amdgcn_s_wait_tensorcnt(0);
        }
#else
        {
            const uint4* src = (const uint4*)(Kb + (long)k0 * Hn);
            uint4* dst = (uint4*)Klds;
            dst[tid * 2]     = src[tid * 2];
            dst[tid * 2 + 1] = src[tid * 2 + 1];
        }
#endif
        // ---- Stage V block transposed: Vlds[h][key] ----
        {
            // thread owns 16 h-contiguous elements of one key row
            const int key   = tid >> 2;
            const int hbase = (tid & 3) * 16;
            union { uint4 q[2]; unsigned short s[16]; } vrow;
            const uint4* vsrc =
                (const uint4*)(Vb + (long)(k0 + key) * Hn + hbase);
            vrow.q[0] = vsrc[0];
            vrow.q[1] = vsrc[1];
#pragma unroll
            for (int i = 0; i < 16; ++i)
                Vlds[(hbase + i) * 32 + key] = vrow.s[i];
        }
        // Prefetch next key block while this one is consumed.
        if (k0 + 32 < Tn) {
            __builtin_prefetch(Kb + (long)(k0 + 32) * Hn, 0, 1);
            __builtin_prefetch(Vb + (long)(k0 + 32) * Hn, 0, 1);
        }
        __syncthreads();

        if (k0 <= qw0 + 15) {                 // wave-uniform causal skip
            // ----- S = Q K^T (two 16x16 tiles, K-dim = 64 in 2 steps) -----
            v8f s[2];
            s[0] = zero8(); s[1] = zero8();
#pragma unroll
            for (int nt = 0; nt < 2; ++nt)
#pragma unroll
                for (int ks = 0; ks < 2; ++ks) {
                    BFrag bk;
                    const uint4* kp = (const uint4*)(Klds +
                        ((nt * 16 + l16) * 64 + ks * 32 + half * 16));
                    bk.q[0] = kp[0];
                    bk.q[1] = kp[1];
                    s[nt] = __builtin_amdgcn_wmma_f32_16x16x32_bf16(
                        false, aQ[ks].v, false, bk.v, (short)0, s[nt],
                        false, false);
                }

            // ----- scale + analytic causal mask -----
#pragma unroll
            for (int nt = 0; nt < 2; ++nt)
#pragma unroll
                for (int r = 0; r < 8; ++r) {
                    int key_g = k0 + nt * 16 + l16;
                    int qrow  = qw0 + r + half * 8;
                    float v = s[nt][r] * inv_scale;
                    s[nt][r] = (key_g > qrow) ? -INFINITY : v;
                }

            // ----- online softmax update -----
#pragma unroll
            for (int r = 0; r < 8; ++r) {
                float t = fmaxf(s[0][r], s[1][r]);
#pragma unroll
                for (int off = 8; off >= 1; off >>= 1)
                    t = fmaxf(t, __shfl_xor(t, off, 32));
                float mnew = fmaxf(mrow[r], t);
                float corr = __expf(mrow[r] - mnew);
                lrow[r] *= corr;
#pragma unroll
                for (int nt = 0; nt < 4; ++nt) o[nt][r] *= corr;
                s[0][r] = __expf(s[0][r] - mnew);
                s[1][r] = __expf(s[1][r] - mnew);
                float ps = s[0][r] + s[1][r];
#pragma unroll
                for (int off = 8; off >= 1; off >>= 1)
                    ps += __shfl_xor(ps, off, 32);
                lrow[r] += ps;
                mrow[r] = mnew;
            }

            // ----- P (C-layout) -> per-wave LDS -> A-layout fragment -----
#pragma unroll
            for (int nt = 0; nt < 2; ++nt)
#pragma unroll
                for (int r = 0; r < 8; ++r)
                    Plds[wave][(r + half * 8) * 32 + nt * 16 + l16] =
                        f2bf(s[nt][r]);
            AFrag pa;                         // same-wave DS ops are in-order
#pragma unroll
            for (int v = 0; v < 8; ++v)
                pa.u[v] = *(const unsigned int*)(&Plds[wave][l16 * 32 +
                                                            a_kbase(v, half)]);

            // ----- O += P V (K-dim = 32 keys) -----
#pragma unroll
            for (int nt = 0; nt < 4; ++nt) {
                BFrag bv;
                const uint4* vp = (const uint4*)(Vlds +
                    ((nt * 16 + l16) * 32 + half * 16));
                bv.q[0] = vp[0];
                bv.q[1] = vp[1];
                o[nt] = __builtin_amdgcn_wmma_f32_16x16x32_bf16(
                    false, pa.v, false, bv.v, (short)0, o[nt], false, false);
            }
        }
    }

    // ----- epilogue: out = O / l -----
#pragma unroll
    for (int nt = 0; nt < 4; ++nt)
#pragma unroll
        for (int r = 0; r < 8; ++r) {
            int qrow = qw0 + r + half * 8;
            out[((long)b * Tn + qrow) * Hn + nt * 16 + l16] =
                o[nt][r] / lrow[r];
        }
}

// ---------------------------------------------------------------------------
// Host launcher. Inputs: x, Wq, Wk, Wv, mask (mask intentionally unused).
// ---------------------------------------------------------------------------
extern "C" void kernel_launch(void* const* d_in, const int* in_sizes, int n_in,
                              void* d_out, int out_size, void* d_ws, size_t ws_size,
                              hipStream_t stream) {
    (void)in_sizes; (void)n_in; (void)out_size; (void)ws_size;

    const float* x  = (const float*)d_in[0];
    const float* Wq = (const float*)d_in[1];
    const float* Wk = (const float*)d_in[2];
    const float* Wv = (const float*)d_in[3];
    float* out = (float*)d_out;

    // Workspace layout (bf16):
    //   [0)            Wt : 3*64*1024      = 393,216 B
    //   [0x60000)      QKV: 3*16384*64*2   = 6,291,456 B
    unsigned short* Wt  = (unsigned short*)d_ws;
    unsigned short* qkv = (unsigned short*)((char*)d_ws + 0x60000);

    const int wtot = 3 * Hn * Dn;
    attn_wcvt_kernel<<<(wtot + 255) / 256, 256, 0, stream>>>(Wq, Wk, Wv, Wt);

    // 16384 rows / (4 waves * 16 rows) = 256 blocks
    attn_qkv_kernel<<<BTn / 64, 128, 0, stream>>>(x, Wt, qkv);

    // 4 batches * 64 q-tiles = 256 blocks
    attn_fa_kernel<<<Bn * (Tn / 64), 128, 0, stream>>>(qkv, out);
}
